// MixtureOfExperts_41944650613264
// MI455X (gfx1250) — compile-verified, matrix operand order checked by reading, not run on hardware
//
#include <hip/hip_runtime.h>
#include <hip/hip_bf16.h>

// ---------------- Problem constants (from the reference) ----------------
#define B_TOK   8192
#define D_DIM   1024
#define H_DIM   2048
#define H2_DIM  1024
#define E_NUM   8
#define O_DIM   10
#define ROWS_TOTAL (2 * B_TOK)   // every token contributes exactly TOP_K=2 rows

typedef __bf16 bf16;
typedef __attribute__((ext_vector_type(16))) __bf16 v16bf;
typedef __attribute__((ext_vector_type(8)))  float  v8f;

// ---- CDNA5 async copy (global -> LDS, ASYNCcnt-tracked), feature-guarded ----
// Probe round 2: builtin exists; param0 = int4* in AS(1) (printed '__device__'),
// param1 = int4* in AS(3), then imm offset, imm cpol.
#if defined(__AMDGCN__) && __has_builtin(__builtin_amdgcn_global_load_async_to_lds_b128)
#define MOE_ASYNC 1
typedef int v4i_ __attribute__((vector_size(16)));
#define AS1(p) ((__attribute__((address_space(1))) v4i_*)(p))
#define AS3(p) ((__attribute__((address_space(3))) v4i_*)(p))
#if __has_builtin(__builtin_amdgcn_s_wait_asynccnt)
#define WAIT_ASYNC(n) __builtin_amdgcn_s_wait_asynccnt(n)
#else
#define WAIT_ASYNC(n) asm volatile("s_wait_asynccnt " #n ::: "memory")
#endif
#else
#define MOE_ASYNC 0
#endif

static __device__ __forceinline__ bf16 f2bf(float f) {
  unsigned u = __builtin_bit_cast(unsigned, f);
  u += 0x7FFFu + ((u >> 16) & 1u);          // round-to-nearest-even
  unsigned short h = (unsigned short)(u >> 16);
  return __builtin_bit_cast(bf16, h);
}
static __device__ __forceinline__ float bf2f(bf16 b) {
  unsigned short s = __builtin_bit_cast(unsigned short, b);
  unsigned u = ((unsigned)s) << 16;
  return __builtin_bit_cast(float, u);
}

// ---------------- init: zero per-expert counters ----------------
__global__ void moe_init(int* __restrict__ cnt) {
  if (threadIdx.x < E_NUM) cnt[threadIdx.x] = 0;
}

// ---------------- elementwise fp32 -> bf16 ----------------
__global__ __launch_bounds__(256) void moe_cvt_bf16(const float* __restrict__ src,
                                                    bf16* __restrict__ dst, int n) {
  int i = blockIdx.x * 256 + threadIdx.x;
  if (i < n) dst[i] = f2bf(src[i]);
}

// ------------- transpose + convert: W[e][R][C] (f32) -> WT[e][C][R] (bf16) -------------
__global__ __launch_bounds__(256) void moe_transpose_bf16(const float* __restrict__ W,
                                                          bf16* __restrict__ WT,
                                                          int R, int C) {
  __shared__ float tile[32][33];
  int e  = blockIdx.z;
  int c0 = blockIdx.x * 32, r0 = blockIdx.y * 32;
  const float* We  = W  + (size_t)e * R * C;
  bf16*        WTe = WT + (size_t)e * R * C;
  int tx = threadIdx.x & 31, ty = threadIdx.x >> 5;   // 32 x 8
  for (int j = ty; j < 32; j += 8)
    tile[j][tx] = We[(size_t)(r0 + j) * C + c0 + tx];
  __syncthreads();
  for (int j = ty; j < 32; j += 8)
    WTe[(size_t)(c0 + j) * R + r0 + tx] = f2bf(tile[tx][j]);
}

// ---------------- router: one wave per token ----------------
// top-2 of softmax == top-2 of logits; renormalized pair weight:
// p0 = 1/(1+e^{l1-l0}), p1 = 1-p0  (global softmax denominator cancels).
__global__ __launch_bounds__(256) void moe_router(const float* __restrict__ x,
                                                  const float* __restrict__ Wr,
                                                  const float* __restrict__ br,
                                                  int* __restrict__ cnt,
                                                  int* __restrict__ topi,
                                                  float* __restrict__ topp) {
  int wave = threadIdx.x >> 5, lane = threadIdx.x & 31;
  int b = blockIdx.x * 8 + wave;
  float acc[E_NUM];
#pragma unroll
  for (int e = 0; e < E_NUM; ++e) acc[e] = 0.f;
  const float* xr = x + (size_t)b * D_DIM;
  for (int d = lane; d < D_DIM; d += 32) {
    float xv = xr[d];
    const float* wr = Wr + (size_t)d * E_NUM;
#pragma unroll
    for (int e = 0; e < E_NUM; ++e) acc[e] += xv * wr[e];
  }
#pragma unroll
  for (int off = 16; off > 0; off >>= 1) {
#pragma unroll
    for (int e = 0; e < E_NUM; ++e) acc[e] += __shfl_xor(acc[e], off);
  }
  if (lane == 0) {
    float l[E_NUM];
#pragma unroll
    for (int e = 0; e < E_NUM; ++e) l[e] = acc[e] + br[e];
    int i0 = 0; float v0 = l[0];
#pragma unroll
    for (int e = 1; e < E_NUM; ++e) if (l[e] > v0) { v0 = l[e]; i0 = e; }
    int i1 = -1; float v1 = -INFINITY;
#pragma unroll
    for (int e = 0; e < E_NUM; ++e) if (e != i0 && l[e] > v1) { v1 = l[e]; i1 = e; }
    float a  = expf(v1 - v0);
    float p0 = 1.f / (1.f + a);
    topi[2 * b] = i0; topi[2 * b + 1] = i1;
    topp[2 * b] = p0; topp[2 * b + 1] = 1.f - p0;
    atomicAdd(&cnt[i0], 1);
    atomicAdd(&cnt[i1], 1);
  }
}

// ---------------- tiny exclusive scan over 8 experts ----------------
__global__ void moe_scan(const int* __restrict__ cnt, int* __restrict__ start,
                         int* __restrict__ cnt2) {
  if (threadIdx.x == 0) {
    int s = 0;
    for (int e = 0; e < E_NUM; ++e) { start[e] = s; s += cnt[e]; cnt2[e] = 0; }
  }
}

// ---------------- scatter tokens into compacted per-expert row ranges ----------------
__global__ __launch_bounds__(256) void moe_scatter(const int* __restrict__ topi,
                                                   const int* __restrict__ start,
                                                   int* __restrict__ cnt2,
                                                   int* __restrict__ tok_row,
                                                   int* __restrict__ exp_row,
                                                   int* __restrict__ row_of) {
  int b = blockIdx.x * 256 + threadIdx.x;
#pragma unroll
  for (int k = 0; k < 2; ++k) {
    int e    = topi[2 * b + k];
    int slot = atomicAdd(&cnt2[e], 1);
    int row  = start[e] + slot;
    tok_row[row]      = b;
    exp_row[row]      = e;
    row_of[2 * b + k] = row;
  }
}

// ---------------- routed GEMM + bias + relu, bf16 WMMA ----------------
// Out[start[e]+m][n] = relu( sum_k A[row_src][k] * BT[e][n][k] + bias[e][n] )
// Block tile 128(M) x 128(N) x 32(K); 8 waves, each wave 32x64 via
// 2x4 v_wmma_f32_16x16x32_bf16. A/B fragments per ISA 7.12.2: lane = M (or N),
// two 8-element K chunks selected by lane[4]; LDS chunks XOR-swizzled.
// When available, staging uses global_load_async_to_lds_b128 with double-
// buffered LDS so the next tile's DMA overlaps the current tile's WMMAs.
__global__ __launch_bounds__(256) void moe_gemm_bf16(const bf16* __restrict__ Abase,
                                                     const bf16* __restrict__ BT,
                                                     const float* __restrict__ bias,
                                                     bf16* __restrict__ Out,
                                                     const int* __restrict__ cnt,
                                                     const int* __restrict__ start,
                                                     const int* __restrict__ tok_row,
                                                     int K, int N, int gather) {
  int e    = blockIdx.z;
  int cntE = cnt[e];
  int m0   = blockIdx.y * 128;
  if (m0 >= cntE) return;                       // routed: skip empty tiles
  int startE = start[e];
  int n0     = blockIdx.x * 128;

#if MOE_ASYNC
  __shared__ __align__(16) bf16 As[2][128 * 32];
  __shared__ __align__(16) bf16 Bs[2][128 * 32];
#else
  __shared__ __align__(16) bf16 As[1][128 * 32];
  __shared__ __align__(16) bf16 Bs[1][128 * 32];
#endif

  const bf16* BTe = BT + (size_t)e * N * K;
  int t    = threadIdx.x;
  int wave = t >> 5, lane = t & 31;
  int mr   = lane & 15, half = lane >> 4;
  int waveM = (wave >> 1) * 32;
  int waveN = (wave & 1) * 64;

  v8f acc[2][4];
  v8f zero = {0.f, 0.f, 0.f, 0.f, 0.f, 0.f, 0.f, 0.f};
#pragma unroll
  for (int i = 0; i < 2; ++i)
#pragma unroll
    for (int j = 0; j < 4; ++j) acc[i][j] = zero;

  // Each thread stages 2 A chunks + 2 B chunks (16B each) per K-step.
  const bf16* aRow[2]; const bf16* bRow[2];
  int ldsOff[2];                       // XOR-swizzled element offset inside a tile
#pragma unroll
  for (int i = 0; i < 2; ++i) {
    int cid = t + i * 256;
    int row = cid >> 2, c = cid & 3;
    ldsOff[i] = row * 32 + ((c ^ (row & 3)) * 8);
    int m = m0 + row;
    int src;
    if (gather) src = (m < cntE) ? tok_row[startE + m] : tok_row[startE];
    else        src = startE + ((m < cntE) ? m : 0);
    aRow[i] = Abase + (size_t)src * K + c * 8;
    bRow[i] = BTe + (size_t)(n0 + row) * K + c * 8;
  }

  int ksteps = K >> 5;

#if MOE_ASYNC
  // Issue one stage = 4 async-to-LDS instructions per wave (retire in order).
#define MOE_STAGE(buf, k0)                                                    \
  {                                                                           \
    _Pragma("unroll")                                                         \
    for (int i = 0; i < 2; ++i) {                                             \
      __builtin_amdgcn_global_load_async_to_lds_b128(                         \
          AS1(aRow[i] + (k0)), AS3(&As[buf][ldsOff[i]]), 0, 0);               \
      __builtin_amdgcn_global_load_async_to_lds_b128(                         \
          AS1(bRow[i] + (k0)), AS3(&Bs[buf][ldsOff[i]]), 0, 0);               \
    }                                                                         \
  }
  MOE_STAGE(0, 0)
#endif

  for (int ks = 0; ks < ksteps; ++ks) {
#if MOE_ASYNC
    int cur = ks & 1;
    if (ks + 1 < ksteps) {
      MOE_STAGE(cur ^ 1, (ks + 1) << 5)   // next tile's DMA overlaps this tile's math
      WAIT_ASYNC(4);                      // oldest 4 (current tile) have landed
    } else {
      WAIT_ASYNC(0);
    }
    __syncthreads();                      // current tile visible to all waves
#else
    int cur = 0;
    int k0 = ks << 5;
    __syncthreads();
#pragma unroll
    for (int i = 0; i < 2; ++i) {
      uint4 av = *(const uint4*)(aRow[i] + k0);
      *(uint4*)(&As[0][ldsOff[i]]) = av;
      uint4 bv = *(const uint4*)(bRow[i] + k0);
      *(uint4*)(&Bs[0][ldsOff[i]]) = bv;
    }
    __syncthreads();
#endif

    v16bf af[2], bfm[4];
#pragma unroll
    for (int sm = 0; sm < 2; ++sm) {
      int row = waveM + sm * 16 + mr;
      union { uint4 u[2]; v16bf v; } f;
      const uint4* base = (const uint4*)(&As[cur][row * 32]);
      f.u[0] = base[(half)     ^ (row & 3)];   // K = half*8 .. +7
      f.u[1] = base[(half + 2) ^ (row & 3)];   // K = 16 + half*8 .. +7
      af[sm] = f.v;
    }
#pragma unroll
    for (int sn = 0; sn < 4; ++sn) {
      int row = waveN + sn * 16 + mr;
      union { uint4 u[2]; v16bf v; } f;
      const uint4* base = (const uint4*)(&Bs[cur][row * 32]);
      f.u[0] = base[(half)     ^ (row & 3)];
      f.u[1] = base[(half + 2) ^ (row & 3)];
      bfm[sn] = f.v;
    }
#pragma unroll
    for (int sm = 0; sm < 2; ++sm)
#pragma unroll
      for (int sn = 0; sn < 4; ++sn)
        acc[sm][sn] = __builtin_amdgcn_wmma_f32_16x16x32_bf16(
            false, af[sm], false, bfm[sn], (short)0, acc[sm][sn], false, false);

    __syncthreads();                      // all reads done before buffer reuse
  }

  int mvalid = cntE - m0; if (mvalid > 128) mvalid = 128;
  const float* biasE = bias + (size_t)e * N;
#pragma unroll
  for (int sm = 0; sm < 2; ++sm) {
#pragma unroll
    for (int sn = 0; sn < 4; ++sn) {
      int nglob = n0 + waveN + sn * 16 + mr;      // C/D: lane -> N
      float bv  = biasE[nglob];
#pragma unroll
      for (int i = 0; i < 8; ++i) {
        int m = waveM + sm * 16 + i + 8 * half;   // C/D: vgpr i -> M (+8 for hi half)
        if (m < mvalid) {
          float v = acc[sm][sn][i] + bv;
          v = v > 0.f ? v : 0.f;
          Out[(size_t)(startE + m0 + m) * N + nglob] = f2bf(v);
        }
      }
    }
  }
}

// ---------------- final projection (N = 10): one wave per routed row ----------------
__global__ __launch_bounds__(256) void moe_out_gemm(const bf16* __restrict__ h2,
                                                    const float* __restrict__ W3,
                                                    const float* __restrict__ b3,
                                                    const int* __restrict__ exp_row,
                                                    float* __restrict__ yrow) {
  int wave = threadIdx.x >> 5, lane = threadIdx.x & 31;
  int row = blockIdx.x * 8 + wave;
  int e   = exp_row[row];
  const bf16*  h = h2 + (size_t)row * H2_DIM;
  const float* w = W3 + (size_t)e * H2_DIM * O_DIM;
  float acc[O_DIM];
#pragma unroll
  for (int o = 0; o < O_DIM; ++o) acc[o] = 0.f;
  for (int k = lane; k < H2_DIM; k += 32) {
    float hv = bf2f(h[k]);
    const float* wr = w + (size_t)k * O_DIM;
#pragma unroll
    for (int o = 0; o < O_DIM; ++o) acc[o] += hv * wr[o];
  }
#pragma unroll
  for (int off = 16; off > 0; off >>= 1) {
#pragma unroll
    for (int o = 0; o < O_DIM; ++o) acc[o] += __shfl_xor(acc[o], off);
  }
  if (lane == 0) {
#pragma unroll
    for (int o = 0; o < O_DIM; ++o)
      yrow[(size_t)row * O_DIM + o] = acc[o] + b3[e * O_DIM + o];
  }
}

// ---------------- combine: out[b] = p0*y[row0] + p1*y[row1] (no fp atomics) ----------------
__global__ __launch_bounds__(256) void moe_combine(const float* __restrict__ yrow,
                                                   const int* __restrict__ row_of,
                                                   const float* __restrict__ topp,
                                                   float* __restrict__ out) {
  int b = blockIdx.x * 256 + threadIdx.x;
  int r0 = row_of[2 * b], r1 = row_of[2 * b + 1];
  float p0 = topp[2 * b], p1 = topp[2 * b + 1];
#pragma unroll
  for (int o = 0; o < O_DIM; ++o)
    out[(size_t)b * O_DIM + o] =
        p0 * yrow[(size_t)r0 * O_DIM + o] + p1 * yrow[(size_t)r1 * O_DIM + o];
}

// ======================================================================
extern "C" void kernel_launch(void* const* d_in, const int* in_sizes, int n_in,
                              void* d_out, int out_size, void* d_ws, size_t ws_size,
                              hipStream_t stream) {
  (void)in_sizes; (void)n_in; (void)out_size; (void)ws_size;
  const float* x  = (const float*)d_in[0];
  const float* Wr = (const float*)d_in[1];
  const float* br = (const float*)d_in[2];
  const float* W1 = (const float*)d_in[3];
  const float* b1 = (const float*)d_in[4];
  const float* W2 = (const float*)d_in[5];
  const float* b2 = (const float*)d_in[6];
  const float* W3 = (const float*)d_in[7];
  const float* b3 = (const float*)d_in[8];
  float* out = (float*)d_out;

  char* ws = (char*)d_ws;
  size_t off = 0;
  auto alloc = [&](size_t bytes) -> char* {
    char* p = ws + off;
    off = (off + bytes + 255) & ~(size_t)255;
    return p;
  };
  int*   cnt     = (int*)  alloc(E_NUM * 4);
  int*   cnt2    = (int*)  alloc(E_NUM * 4);
  int*   start   = (int*)  alloc(E_NUM * 4);
  int*   topi    = (int*)  alloc((size_t)B_TOK * 2 * 4);
  float* topp    = (float*)alloc((size_t)B_TOK * 2 * 4);
  int*   row_of  = (int*)  alloc((size_t)B_TOK * 2 * 4);
  int*   tok_row = (int*)  alloc((size_t)ROWS_TOTAL * 4);
  int*   exp_row = (int*)  alloc((size_t)ROWS_TOTAL * 4);
  float* yrow    = (float*)alloc((size_t)ROWS_TOTAL * O_DIM * 4);
  bf16*  Xbf     = (bf16*) alloc((size_t)B_TOK * D_DIM * 2);
  bf16*  W1T     = (bf16*) alloc((size_t)E_NUM * D_DIM * H_DIM * 2);
  bf16*  W2T     = (bf16*) alloc((size_t)E_NUM * H_DIM * H2_DIM * 2);
  bf16*  h1bf    = (bf16*) alloc((size_t)ROWS_TOTAL * H_DIM * 2);
  bf16*  h2bf    = (bf16*) alloc((size_t)ROWS_TOTAL * H2_DIM * 2);

  moe_init<<<1, 32, 0, stream>>>(cnt);
  moe_cvt_bf16<<<(B_TOK * D_DIM) / 256, 256, 0, stream>>>(x, Xbf, B_TOK * D_DIM);
  // W1[e][D][H] -> W1T[e][H][D],  W2[e][H][H2] -> W2T[e][H2][H]
  moe_transpose_bf16<<<dim3(H_DIM / 32, D_DIM / 32, E_NUM), 256, 0, stream>>>(W1, W1T, D_DIM, H_DIM);
  moe_transpose_bf16<<<dim3(H2_DIM / 32, H_DIM / 32, E_NUM), 256, 0, stream>>>(W2, W2T, H_DIM, H2_DIM);
  moe_router<<<B_TOK / 8, 256, 0, stream>>>(x, Wr, br, cnt, topi, topp);
  moe_scan<<<1, 32, 0, stream>>>(cnt, start, cnt2);
  moe_scatter<<<B_TOK / 256, 256, 0, stream>>>(topi, start, cnt2, tok_row, exp_row, row_of);
  // stage 1: h1 = relu(Xg @ W1 + b1)   (gathered tokens, K=1024, N=2048)
  moe_gemm_bf16<<<dim3(H_DIM / 128, B_TOK / 128, E_NUM), 256, 0, stream>>>(
      Xbf, W1T, b1, h1bf, cnt, start, tok_row, D_DIM, H_DIM, 1);
  // stage 2: h2 = relu(h1 @ W2 + b2)   (compacted rows, K=2048, N=1024)
  moe_gemm_bf16<<<dim3(H2_DIM / 128, B_TOK / 128, E_NUM), 256, 0, stream>>>(
      h1bf, W2T, b2, h2bf, cnt, start, nullptr, H_DIM, H2_DIM, 0);
  moe_out_gemm<<<ROWS_TOTAL / 8, 256, 0, stream>>>(h2bf, W3, b3, exp_row, yrow);
  moe_combine<<<B_TOK / 256, 256, 0, stream>>>(yrow, row_of, topp, out);
}
